// DEFLocalTransBottleneck_22402549416245
// MI455X (gfx1250) — matrix operand order repaired and checked by reading, not compiled, not run on hardware
//
#include <hip/hip_runtime.h>
#include <hip/hip_bf16.h>
#include <math.h>

// ---------------- problem constants (from reference) ----------------
#define cN   4
#define cCIN 1024
#define cCW  256
#define cT   8
#define cH   28
#define cW   28
#define cHW  (cH * cW)          // 784
#define cP   (cT * cHW)         // 6272 pixels per (n, channel)
#define cKK  9
#define cEPS 1e-5f

// ---------------- vector types ----------------
typedef __attribute__((ext_vector_type(16))) __bf16 v16bf;
typedef __attribute__((ext_vector_type(8)))  float  v8f;
typedef __attribute__((ext_vector_type(4)))  unsigned int u32x4;
typedef __attribute__((ext_vector_type(2)))  unsigned int u32x2;
typedef __attribute__((ext_vector_type(4)))  float  f32x4;

// GEMM tiling: block = 256 threads = 8 waves; block tile 64(M) x 128(pix);
// each wave computes a 32x32 tile = 4 WMMA accumulators per K-step.
#define BM  64
#define BNP 128
#define BK  32
#define LDP 40   // LDS row pitch in halves (80B: 16B-aligned, bank-conflict-free)

// =====================================================================
// Weight conversion (fp32 -> bf16; optional [O][I][9] -> [9][O][I])
// =====================================================================
__global__ void cvt_mat_kernel(const float* __restrict__ src,
                               __bf16* __restrict__ dst, int n) {
  int i = blockIdx.x * blockDim.x + threadIdx.x;
  if (i < n) dst[i] = (__bf16)src[i];
}

__global__ void cvt_tap_kernel(const float* __restrict__ src,  // [O][I][9]
                               __bf16* __restrict__ dst,       // [9][O][I]
                               int O, int I) {
  int i = blockIdx.x * blockDim.x + threadIdx.x;
  int n = O * I * cKK;
  if (i >= n) return;
  int o   = i / (I * cKK);
  int r   = i - o * (I * cKK);
  int ic  = r / cKK;
  int tap = r - ic * cKK;
  dst[tap * O * I + o * I + ic] = (__bf16)src[i];
}

// =====================================================================
// WMMA GEMM, double-buffered LDS, split-phase staging:
//   loop body: barrier -> [global loads for tile i+1] -> frag ds loads ->
//              4x WMMA (waits only frag loads) -> cvt + ds stores (i+1)
// BTRANS=true: B stored pixel-major [n][p][k] (no taps).
// =====================================================================
template<int TAPS, bool BN_EN, bool ADD_ID, bool RELU_EN, bool BTRANS>
__global__ __launch_bounds__(256) void gemm_wmma_kernel(
    const __bf16* __restrict__ A,     // [TAPS][Mvalid][Kc]
    const float*  __restrict__ Bact,  // [N][Kc][cP]  or  [N][cP][Kc]
    const float*  __restrict__ bnp,   // [4][Mvalid] or nullptr
    const float*  __restrict__ ident, // [N][Mvalid][cP] or nullptr
    float*        __restrict__ Out,   // [N][Mvalid][cP]
    int Mvalid, int Kc)
{
  __shared__ __bf16 As[2][BM][LDP];
  __shared__ __bf16 Bs[2][BNP][LDP];

  const int nbat = blockIdx.z;
  const int m0   = blockIdx.y * BM;
  const int p0   = blockIdx.x * BNP;
  const int tid  = threadIdx.x;
  const int lane = tid & 31;
  const int wave = tid >> 5;     // 0..7
  const int wn   = wave & 3;     // n-tile pair 0..3
  const int wm2  = wave >> 2;    // m half 0..1 (rows wm2*32 .. +31)

  // batch-adjusted base pointer; all further indices are 32-bit
  const float* Bb = Bact + (size_t)nbat * Kc * cP;

  const int KIT = Kc / BK;
  const int TOT = TAPS * KIT;

  v8f acc00 = {}, acc01 = {}, acc10 = {}, acc11 = {};

  // staging registers (live across the WMMAs of the previous tile)
  u32x4 aD = {0u, 0u, 0u, 0u};
  bool  aval = false;
  float bv[4][4];
  float bf[4];

  // ---- phase 1: global loads into registers (branch-free) ----
  auto stage_load = [&](int it) {
    int tap = 0, k0;
    if (TAPS > 1) { tap = it / KIT; k0 = (it - tap * KIT) * BK; }
    else          { k0 = it * BK; }
    const int dy = tap / 3 - 1, dx = tap % 3 - 1;
    const __bf16* Atap = A + tap * Mvalid * Kc;

    // A: 64 rows x 32 k, 16B per thread
    {
      int r  = tid >> 2;
      int c8 = (tid & 3) * 8;
      int gm = m0 + r;
      aval = (gm < Mvalid);
      int aidx = (aval ? gm : (Mvalid - 1)) * Kc + k0 + c8;
      aD = *(const u32x4*)(Atap + aidx);
    }
    // B: 32 k x 128 pix; 4 k per thread
    #pragma unroll
    for (int i = 0; i < 4; ++i) {
      int lin = tid + i * 256;            // 0..1023
      if (BTRANS) {
        int kq = lin & 7;                 // k quad 0..7
        int px = lin >> 3;                // 0..127
        f32x4 d = *(const f32x4*)&Bb[(p0 + px) * Kc + k0 + kq * 4];
        bv[i][0] = d.x; bv[i][1] = d.y; bv[i][2] = d.z; bv[i][3] = d.w;
        bf[i] = 1.0f;
      } else {
        int kq = lin >> 7;                // 0..7
        int px = lin & 127;
        int p = p0 + px;
        int k = k0 + kq * 4;
        int ps = p;
        float f = 1.0f;
        if (TAPS > 1) {
          int rem = p % cHW;
          int h_ = rem / cW, w_ = rem - h_ * cW;
          int hh = h_ + dy, ww = w_ + dx;
          bool valid = (hh >= 0) & (hh < cH) & (ww >= 0) & (ww < cW);
          ps = valid ? (p + dy * cW + dx) : p;   // clamped: always in-bounds
          f  = valid ? 1.0f : 0.0f;
        }
        int bidx = k * cP + ps;
        #pragma unroll
        for (int j = 0; j < 4; ++j) bv[i][j] = Bb[bidx + j * cP];
        bf[i] = f;
      }
    }
    // prefetch a future K-slab (global_prefetch_b8)
    if (TAPS == 1 && !BTRANS && it + 2 < TOT)
      __builtin_prefetch(&Bb[((it + 2) * BK + (tid >> 7)) * cP + p0 + (tid & 127)], 0, 0);
  };

  // ---- phase 2: convert + LDS stores ----
  auto stage_store = [&](int buf) {
    {
      u32x4 z = {0u, 0u, 0u, 0u};
      u32x4 d = aval ? aD : z;
      *(u32x4*)&As[buf][tid >> 2][(tid & 3) * 8] = d;
    }
    #pragma unroll
    for (int i = 0; i < 4; ++i) {
      int lin = tid + i * 256;
      int kq, px;
      if (BTRANS) { kq = lin & 7;  px = lin >> 3; }
      else        { kq = lin >> 7; px = lin & 127; }
      __bf16 h[4];
      #pragma unroll
      for (int j = 0; j < 4; ++j) h[j] = (__bf16)(bv[i][j] * bf[i]);
      *(u32x2*)&Bs[buf][px][kq * 4] = *(u32x2*)h;
    }
  };

  stage_load(0);
  stage_store(0);

  for (int it = 0; it < TOT; ++it) {
    const int buf = it & 1;
    __syncthreads();

    // issue next tile's global loads first (long latency)
    if (it + 1 < TOT) stage_load(it + 1);

    // ---- fragment ds loads (only outstanding DS ops at this point) ----
    const int r  = lane & 15;
    const int kg = lane >> 4;
    v16bf a0, a1;
    {
      const __bf16* Ar0 = &As[buf][wm2 * 32 + r][0];
      const __bf16* Ar1 = &As[buf][wm2 * 32 + 16 + r][0];
      #pragma unroll
      for (int i = 0; i < 8; ++i) {
        a0[i]     = Ar0[kg * 8 + i];
        a0[8 + i] = Ar0[16 + kg * 8 + i];
        a1[i]     = Ar1[kg * 8 + i];
        a1[8 + i] = Ar1[16 + kg * 8 + i];
      }
    }
    v16bf b0, b1;
    {
      const int col = lane & 15, kb = (lane >> 4) * 8;
      const __bf16* B0 = &Bs[buf][(2 * wn) * 16 + col][0];
      const __bf16* B1 = &Bs[buf][(2 * wn + 1) * 16 + col][0];
      #pragma unroll
      for (int i = 0; i < 8; ++i) {
        b0[i] = B0[kb + i]; b0[8 + i] = B0[16 + kb + i];
        b1[i] = B1[kb + i]; b1[8 + i] = B1[16 + kb + i];
      }
    }
    acc00 = __builtin_amdgcn_wmma_f32_16x16x32_bf16(false, a0, false, b0, (short)0, acc00, false, false);
    acc01 = __builtin_amdgcn_wmma_f32_16x16x32_bf16(false, a0, false, b1, (short)0, acc01, false, false);
    acc10 = __builtin_amdgcn_wmma_f32_16x16x32_bf16(false, a1, false, b0, (short)0, acc10, false, false);
    acc11 = __builtin_amdgcn_wmma_f32_16x16x32_bf16(false, a1, false, b1, (short)0, acc11, false, false);

    // now drain the global loads and fill the other LDS buffer
    if (it + 1 < TOT) stage_store(buf ^ 1);
  }

  // ---- epilogue: BN / +identity / ReLU, fp32 store ----
  const int colb = lane & 15, rsel = lane >> 4;
  #pragma unroll
  for (int mi = 0; mi < 2; ++mi) {
    #pragma unroll
    for (int j = 0; j < 8; ++j) {
      int gm = m0 + wm2 * 32 + mi * 16 + rsel * 8 + j;
      if (gm >= Mvalid) continue;
      float sc = 1.0f, sh = 0.0f;
      if (BN_EN) {
        float g  = bnp[gm];
        float b  = bnp[Mvalid + gm];
        float mu = bnp[2 * Mvalid + gm];
        float vr = bnp[3 * Mvalid + gm];
        sc = g * rsqrtf(vr + cEPS);
        sh = b - mu * sc;
      }
      int rowoff = nbat * Mvalid * cP + gm * cP;
      #pragma unroll
      for (int s = 0; s < 2; ++s) {
        float av;
        if (mi == 0) av = (s == 0) ? acc00[j] : acc01[j];
        else         av = (s == 0) ? acc10[j] : acc11[j];
        int p = p0 + (2 * wn + s) * 16 + colb;
        float v = av * sc + sh;
        if (ADD_ID) v += ident[rowoff + p];
        if (RELU_EN) v = fmaxf(v, 0.0f);
        Out[rowoff + p] = v;
      }
    }
  }
}

// =====================================================================
// Temporal-shift sigmoid gating: out_tmp = sigmoid(out - prev)*out + out
// =====================================================================
__global__ void gating_kernel(const float* __restrict__ out2,
                              float* __restrict__ out_tmp) {
  int idx = blockIdx.x * blockDim.x + threadIdx.x;   // < 6.4M: 32-bit safe
  if (idx >= cN * cCW * cP) return;
  int pin = idx % cP;
  int t = pin / cHW;
  float v  = out2[idx];
  float pv = (t > 0) ? out2[idx - cHW] : v;
  float s  = 1.0f / (1.0f + expf(-(v - pv)));
  out_tmp[idx] = s * v + v;
}

// =====================================================================
// Tiled transpose  [N][C][P] -> [N][P][C]   (C = 256)
// =====================================================================
__global__ __launch_bounds__(256) void transpose_kernel(
    const float* __restrict__ src, float* __restrict__ dst) {
  __shared__ float tile[32][33];
  const int n  = blockIdx.z;
  const int p0 = blockIdx.x * 32;
  const int c0 = blockIdx.y * 32;
  const int tx = threadIdx.x & 31;
  const int ty = threadIdx.x >> 5;   // 0..7
  const float* s = src + (size_t)n * cCW * cP;
  float*       d = dst + (size_t)n * cP * cCW;
  #pragma unroll
  for (int j = 0; j < 4; ++j)
    tile[ty + 8 * j][tx] = s[(c0 + ty + 8 * j) * cP + p0 + tx];
  __syncthreads();
  #pragma unroll
  for (int j = 0; j < 4; ++j)
    d[(p0 + ty + 8 * j) * cCW + c0 + tx] = tile[tx][ty + 8 * j];
}

// =====================================================================
// Fused deform sampling + correlation + aggregation + frame shift.
// Pixel-major feature layout: coalesced float4 channel access.
//   pre3t[n][p(t)][:] = out2t[n][p(t)][:] + agg[n][:,t+1]   (t < T-1)
// One wave per output pixel; 8 consecutive channels per lane.
// =====================================================================
__global__ __launch_bounds__(256) void deform_agg_kernel(
    const float* __restrict__ out2t, // [N][cP][Cw]
    const float* __restrict__ offs,  // [N][18][cP]
    float* __restrict__ pre3t)       // [N][cP][Cw]
{
  const int lane = threadIdx.x & 31;
  const int wid  = blockIdx.x * (blockDim.x >> 5) + (threadIdx.x >> 5);
  if (wid >= cN * cP) return;

  const int n   = wid / cP;
  const int rem = wid - n * cP;
  const int t   = rem / cHW;
  const int hw  = rem - t * cHW;
  const int h   = hw / cW;
  const int w   = hw - h * cW;

  const float* outtn = out2t + (size_t)n * cP * cCW;
  float*       pren  = pre3t + (size_t)n * cP * cCW;
  const int pidx = t * cHW + hw;
  const int cb   = lane * 8;          // this lane's 8 channels

  f32x4 base0 = *(const f32x4*)&outtn[pidx * cCW + cb];
  f32x4 base1 = *(const f32x4*)&outtn[pidx * cCW + cb + 4];

  if (t == cT - 1) {
    *(f32x4*)&pren[pidx * cCW + cb]     = base0;
    *(f32x4*)&pren[pidx * cCW + cb + 4] = base1;
    return;
  }

  const int ti = t + 1;                            // offsets + corr frame
  const int tf = (ti + 1 < cT) ? ti + 1 : cT - 1;  // feat_next frame
  const float* offp = offs + (size_t)n * (2 * cKK) * cP + ti * cHW + hw;

  f32x4 oc0 = *(const f32x4*)&outtn[(ti * cHW + hw) * cCW + cb];
  f32x4 oc1 = *(const f32x4*)&outtn[(ti * cHW + hw) * cCW + cb + 4];

  float agg[8] = {0.f, 0.f, 0.f, 0.f, 0.f, 0.f, 0.f, 0.f};

  #pragma unroll
  for (int kk = 0; kk < cKK; ++kk) {
    float py = (float)(h + (kk / 3) - 1) + offp[(2 * kk) * cP];
    float px = (float)(w + (kk % 3) - 1) + offp[(2 * kk + 1) * cP];
    float y0f = floorf(py), x0f = floorf(px);
    float wy1 = py - y0f, wx1 = px - x0f;
    int y0 = (int)y0f, x0 = (int)x0f;

    float samp[8] = {0.f, 0.f, 0.f, 0.f, 0.f, 0.f, 0.f, 0.f};
    #pragma unroll
    for (int dy = 0; dy < 2; ++dy) {
      #pragma unroll
      for (int dx = 0; dx < 2; ++dx) {
        int yy = y0 + dy, xx = x0 + dx;
        bool valid = (yy >= 0) & (yy < cH) & (xx >= 0) & (xx < cW);
        float wgt = (dy ? wy1 : 1.0f - wy1) * (dx ? wx1 : 1.0f - wx1);
        wgt = valid ? wgt : 0.0f;
        int sidx = valid ? (yy * cW + xx) : 0;     // clamped: always in-bounds
        const float* fr = &outtn[(tf * cHW + sidx) * cCW + cb];
        f32x4 g0 = *(const f32x4*)fr;
        f32x4 g1 = *(const f32x4*)(fr + 4);
        samp[0] += wgt * g0.x; samp[1] += wgt * g0.y;
        samp[2] += wgt * g0.z; samp[3] += wgt * g0.w;
        samp[4] += wgt * g1.x; samp[5] += wgt * g1.y;
        samp[6] += wgt * g1.z; samp[7] += wgt * g1.w;
      }
    }
    // channel-normalized correlation: wave32 reduction
    float part = oc0.x * samp[0] + oc0.y * samp[1] + oc0.z * samp[2] + oc0.w * samp[3]
               + oc1.x * samp[4] + oc1.y * samp[5] + oc1.z * samp[6] + oc1.w * samp[7];
    #pragma unroll
    for (int m = 16; m > 0; m >>= 1) part += __shfl_xor(part, m, 32);
    float corr = part * (1.0f / (float)cCW);
    #pragma unroll
    for (int i = 0; i < 8; ++i) agg[i] += corr * samp[i];
  }

  f32x4 r0 = {base0.x + agg[0], base0.y + agg[1], base0.z + agg[2], base0.w + agg[3]};
  f32x4 r1 = {base1.x + agg[4], base1.y + agg[5], base1.z + agg[6], base1.w + agg[7]};
  *(f32x4*)&pren[pidx * cCW + cb]     = r0;
  *(f32x4*)&pren[pidx * cCW + cb + 4] = r1;
}

// =====================================================================
// Launch
// =====================================================================
extern "C" void kernel_launch(void* const* d_in, const int* in_sizes, int n_in,
                              void* d_out, int out_size, void* d_ws, size_t ws_size,
                              hipStream_t stream) {
  (void)in_sizes; (void)n_in; (void)out_size; (void)ws_size;

  const float* x      = (const float*)d_in[0];  // [N,1024,T,H,W]
  const float* w1     = (const float*)d_in[1];  // [256,1024]
  const float* bn1    = (const float*)d_in[2];  // [4,256]
  const float* w2     = (const float*)d_in[3];  // [256,256,3,3]
  const float* bn2    = (const float*)d_in[4];  // [4,256]
  const float* woff   = (const float*)d_in[5];  // [18,256,3,3]
  const float* w3     = (const float*)d_in[6];  // [1024,256]
  const float* bn3    = (const float*)d_in[7];  // [4,1024]
  float* out          = (float*)d_out;          // [N,1024,T,H,W]

  // ---- workspace layout ----
  char* ws = (char*)d_ws;
  size_t off = 0;
  auto alloc = [&](size_t bytes) {
    void* p = ws + off;
    off += (bytes + 255) & ~(size_t)255;
    return p;
  };
  const size_t actF = (size_t)cN * cCW * cP;
  float*  out1    = (float*)alloc(actF * 4);
  float*  out2    = (float*)alloc(actF * 4);
  float*  out_tmp = (float*)alloc(actF * 4);
  float*  out2t   = (float*)alloc(actF * 4);   // [N][P][C]
  float*  pre3t   = (float*)alloc(actF * 4);   // [N][P][C]
  float*  offbuf  = (float*)alloc((size_t)cN * 2 * cKK * cP * 4);
  __bf16* w1b     = (__bf16*)alloc((size_t)cCW * cCIN * 2);
  __bf16* w2t     = (__bf16*)alloc((size_t)cKK * cCW * cCW * 2);
  __bf16* wofft   = (__bf16*)alloc((size_t)cKK * 2 * cKK * cCW * 2);
  __bf16* w3b     = (__bf16*)alloc((size_t)cCIN * cCW * 2);

  // ---- weight conversion ----
  {
    int n1 = cCW * cCIN;
    cvt_mat_kernel<<<(n1 + 255) / 256, 256, 0, stream>>>(w1, w1b, n1);
    int n3 = cCIN * cCW;
    cvt_mat_kernel<<<(n3 + 255) / 256, 256, 0, stream>>>(w3, w3b, n3);
    int n2 = cCW * cCW * cKK;
    cvt_tap_kernel<<<(n2 + 255) / 256, 256, 0, stream>>>(w2, w2t, cCW, cCW);
    int no = 2 * cKK * cCW * cKK;
    cvt_tap_kernel<<<(no + 255) / 256, 256, 0, stream>>>(woff, wofft, 2 * cKK, cCW);
  }

  const dim3 blk(256);
  const int gx = cP / BNP;  // 49

  // conv1 (1x1, 1024->256) + BN1 + ReLU
  gemm_wmma_kernel<1, true, false, true, false>
      <<<dim3(gx, cCW / BM, cN), blk, 0, stream>>>(w1b, x, bn1, nullptr, out1, cCW, cCIN);

  // conv2 (3x3, 256->256) + BN2 + ReLU
  gemm_wmma_kernel<cKK, true, false, true, false>
      <<<dim3(gx, cCW / BM, cN), blk, 0, stream>>>(w2t, out1, bn2, nullptr, out2, cCW, cCW);

  // temporal gating
  gating_kernel<<<(cN * cCW * cP + 255) / 256, blk, 0, stream>>>(out2, out_tmp);

  // conv_offset (3x3, 256->18), raw output
  gemm_wmma_kernel<cKK, false, false, false, false>
      <<<dim3(gx, 1, cN), blk, 0, stream>>>(wofft, out_tmp, nullptr, nullptr, offbuf, 2 * cKK, cCW);

  // transpose out2 -> pixel-major for the gather stage
  transpose_kernel<<<dim3(cP / 32, cCW / 32, cN), blk, 0, stream>>>(out2, out2t);

  // deform sample + correlation + aggregation + frame shift + residual add
  {
    int waves = cN * cP;  // one wave per pixel
    deform_agg_kernel<<<(waves + 7) / 8, blk, 0, stream>>>(out2t, offbuf, pre3t);
  }

  // conv3 (1x1, 256->1024) + BN3 + identity + ReLU -> d_out  (B transposed)
  gemm_wmma_kernel<1, true, true, true, true>
      <<<dim3(gx, cCIN / BM, cN), blk, 0, stream>>>(w3b, pre3t, bn3, x, out, cCIN, cCW);
}
